// ResGit_18107582120183
// MI455X (gfx1250) — compile-verified, matrix operand order checked by reading, not run on hardware
//
#include <hip/hip_runtime.h>
#include <hip/hip_bf16.h>

typedef __attribute__((ext_vector_type(16))) _Float16 v16h;
typedef __attribute__((ext_vector_type(8)))  float    v8f;

// ---- problem constants (from reference) ----
constexpr int B_   = 128;
constexpr int I_   = 800;
constexpr int E_   = 512;
constexpr int V_   = 19000;
constexpr int NH_  = 8;
constexpr int NB_  = 4;
constexpr int OUT_ = 5000;
constexpr int NPAD_ = 5056;            // 79 * 64, zero-padded final-GEMM N
constexpr int APITCH_ = 520;           // padded LDS row pitch (halves) -> conflict-free b128
constexpr float MASKV_ = -1e9f;

// ---- optional CDNA5 async global->LDS path (guarded; safe fallback) ----
#if defined(__has_builtin)
#if __has_builtin(__builtin_amdgcn_global_load_async_to_lds_b128) && \
    __has_builtin(__builtin_amdgcn_s_wait_asynccnt)
#define USE_ASYNC_LDS 1
#endif
#endif

#ifdef USE_ASYNC_LDS
// builtin parameter type revealed by clang diagnostic: int __vector(4) *
typedef int async_v4i __attribute__((vector_size(16)));
#endif

// =====================================================================
// WMMA fragment helpers (layouts per CDNA5 ISA §7.12.2)
// =====================================================================
__device__ __forceinline__ int kpair_off(int lane, int j) {
  const int kb = (lane & 16) ? 8 : 0;                 // lanes 16-31 hold K+8 / K+24
  return (j < 4) ? (kb + 2 * j) : (16 + kb + 2 * (j - 4));
}

// A fragment (16x32 f16) from an fp32 row-major source; rowbase = this lane's row.
__device__ __forceinline__ v16h load_a_f32(const float* rowbase, int k0, int lane) {
  v16h a;
#pragma unroll
  for (int j = 0; j < 8; ++j) {
    const int k = k0 + kpair_off(lane, j);
    const float2 f = *reinterpret_cast<const float2*>(rowbase + k);
    a[2 * j]     = (_Float16)f.x;
    a[2 * j + 1] = (_Float16)f.y;
  }
  return a;
}

// A or B fragment from f16 data contiguous along K (base = this lane's row/col).
__device__ __forceinline__ v16h load_f16pairs(const _Float16* base, int k0, int lane) {
  v16h b;
#pragma unroll
  for (int j = 0; j < 8; ++j) {
    const int k = k0 + kpair_off(lane, j);            // k is even -> 4B aligned
    union { unsigned u; _Float16 h[2]; } t;
    t.u = *reinterpret_cast<const unsigned*>(base + k);
    b[2 * j]     = t.h[0];
    b[2 * j + 1] = t.h[1];
  }
  return b;
}

__device__ __forceinline__ v8f wmma_f16f32(v16h a, v16h b, v8f c) {
  // (neg_a, A, neg_b, B, c_mod, C, reuse_a, reuse_b)
  return __builtin_amdgcn_wmma_f32_16x16x32_f16(false, a, false, b, (short)0, c,
                                                false, false);
}

// =====================================================================
// Prep kernels: transpose + fp32->fp16 conversion of weights into scratch
// =====================================================================
__global__ void prep_wkT(const float* __restrict__ Wk, _Float16* __restrict__ WkT) {
  const int idx = blockIdx.x * blockDim.x + threadIdx.x;
  if (idx >= NB_ * E_ * E_) return;
  const int blk = idx / (E_ * E_);
  const int rem = idx - blk * E_ * E_;
  const int k = rem / E_, n = rem - k * E_;            // Wk[blk][k][n]
  WkT[((size_t)blk * E_ + n) * E_ + k] = (_Float16)Wk[idx];
}

__global__ void prep_whT(const float* __restrict__ Wh, _Float16* __restrict__ WhT) {
  const int idx = blockIdx.x * blockDim.x + threadIdx.x;
  if (idx >= (NB_ - 1) * E_ * E_) return;
  const int blk = idx / (E_ * E_);
  const int rem = idx - blk * E_ * E_;
  const int k = rem / E_, n = rem - k * E_;
  WhT[((size_t)blk * E_ + n) * E_ + k] = (_Float16)Wh[idx];
}

__global__ void prep_wfT(const float* __restrict__ Wf, _Float16* __restrict__ WfT) {
  const int idx = blockIdx.x * blockDim.x + threadIdx.x;
  if (idx >= NPAD_ * E_) return;
  const int n = idx / E_, k = idx - n * E_;
  WfT[(size_t)n * E_ + k] =
      (n < OUT_) ? (_Float16)Wf[(size_t)k * OUT_ + n] : (_Float16)0.0f;
}

__global__ void prep_wqT(const float* __restrict__ Wq, _Float16* __restrict__ WqT) {
  const int idx = blockIdx.x * blockDim.x + threadIdx.x;
  if (idx >= NB_ * 16 * E_) return;
  const int blk = idx / (16 * E_);
  const int rem = idx - blk * 16 * E_;
  const int h = rem / E_, n = rem - h * E_;
  WqT[idx] = (h < NH_) ? (_Float16)Wq[((size_t)blk * E_ + n) * NH_ + h]
                       : (_Float16)0.0f;
}

// =====================================================================
// scores_kernel: fused  tanh(gather(emb) @ Wk + bk) @ Wq + bq  (+ masking)
//   grid (B*I/64, 1, NB), block 128 (4 waves x 16 rows, full N=512 per WG)
//   A (gathered rows) staged once in LDS as f16; B (WkT slab) staged per
//   N-chunk via async global->LDS; inner loop is ds_load_b128 -> v_wmma only.
// =====================================================================
__global__ void __launch_bounds__(128)
scores_kernel(const int* __restrict__ sga, const float* __restrict__ emb_tables,
              const _Float16* __restrict__ WkT, const float* __restrict__ bk,
              const _Float16* __restrict__ WqT, const float* __restrict__ bq,
              float* __restrict__ scores) {
  const int tid  = threadIdx.x;
  const int lane = tid & 31;
  const int wave = tid >> 5;
  const int blk  = blockIdx.z;
  const int wg_row0 = blockIdx.x * 64;                 // 102400 rows total
  const int row0 = wg_row0 + wave * 16;
  const int nl   = lane & 15;
  const int rbase = (lane & 16) ? 8 : 0;

  __shared__ _Float16 abuf[64][APITCH_];               // gathered A rows (f16)
  __shared__ _Float16 bbuf[64][APITCH_];               // WkT slab (f16)
  __shared__ _Float16 kbuf[4][16][72];                 // per-wave key tile
  __shared__ int sgene[64];

  if (tid < 64) sgene[tid] = sga[wg_row0 + tid];
  __syncthreads();

  // ---- stage A: 64 gathered fp32 rows -> f16 LDS (once per WG) ----
  const float* tab = emb_tables + (size_t)blk * (V_ + 1) * E_;
  for (int u = tid; u < 64 * 256; u += 128) {          // 256 float2 per row
    const int r  = u >> 8;
    const int c2 = (u & 255) * 2;
    const float2 f =
        *reinterpret_cast<const float2*>(tab + (size_t)sgene[r] * E_ + c2);
    union { unsigned u32; _Float16 h[2]; } p;
    p.h[0] = (_Float16)f.x;
    p.h[1] = (_Float16)f.y;
    *reinterpret_cast<unsigned*>(&abuf[r][c2]) = p.u32;
  }
  __syncthreads();

  const _Float16* arow = &abuf[wave * 16 + nl][0];
  const float* bkb = bk + (size_t)blk * E_;
  v8f sacc = {};                                       // 16x16, cols 0..7 = heads

  for (int nc = 0; nc < 8; ++nc) {                     // N chunks of 64
    const int n0 = nc * 64;

    // ---- stage B: WkT[blk][n0..n0+64) x 512 halves -> LDS (shared by waves)
    const _Float16* bsrc = WkT + ((size_t)blk * E_ + n0) * E_;
    for (int u = tid; u < 4096; u += 128) {            // 16B transfers
      const int col = u >> 6;
      const int off = (u & 63) * 8;
#ifdef USE_ASYNC_LDS
      __builtin_amdgcn_global_load_async_to_lds_b128(
          (async_v4i*)(bsrc + (size_t)col * E_ + off),
          (async_v4i*)(&bbuf[col][off]), 0, 0);
#else
      *reinterpret_cast<float4*>(&bbuf[col][off]) =
          *reinterpret_cast<const float4*>(bsrc + (size_t)col * E_ + off);
#endif
    }
#ifdef USE_ASYNC_LDS
    __builtin_amdgcn_s_wait_asynccnt(0);
#endif
    __syncthreads();

    v8f acc0 = {}, acc1 = {}, acc2 = {}, acc3 = {};
    const _Float16* cb0 = &bbuf[nl][0];
    const _Float16* cb1 = &bbuf[16 + nl][0];
    const _Float16* cb2 = &bbuf[32 + nl][0];
    const _Float16* cb3 = &bbuf[48 + nl][0];

    for (int ks = 0; ks < 16; ++ks) {                  // K = 512
      const int k0 = ks * 32;
      const v16h a = load_f16pairs(arow, k0, lane);
      acc0 = wmma_f16f32(a, load_f16pairs(cb0, k0, lane), acc0);
      acc1 = wmma_f16f32(a, load_f16pairs(cb1, k0, lane), acc1);
      acc2 = wmma_f16f32(a, load_f16pairs(cb2, k0, lane), acc2);
      acc3 = wmma_f16f32(a, load_f16pairs(cb3, k0, lane), acc3);
    }

    // epilogue: keys = tanh(acc + bk), stash f16 tile in LDS (C-layout -> rows)
#pragma unroll
    for (int r = 0; r < 8; ++r) {
      const int m = r + rbase;
      kbuf[wave][m][nl]      = (_Float16)tanhf(acc0[r] + bkb[n0 + nl]);
      kbuf[wave][m][16 + nl] = (_Float16)tanhf(acc1[r] + bkb[n0 + 16 + nl]);
      kbuf[wave][m][32 + nl] = (_Float16)tanhf(acc2[r] + bkb[n0 + 32 + nl]);
      kbuf[wave][m][48 + nl] = (_Float16)tanhf(acc3[r] + bkb[n0 + 48 + nl]);
    }
    __syncthreads();

    // scores += keys(16x64) @ Wq(64x16padded) : two more WMMAs
    const _Float16* krow = &kbuf[wave][nl][0];
    const _Float16* qcol = WqT + ((size_t)blk * 16 + nl) * E_ + n0;
    sacc = wmma_f16f32(load_f16pairs(krow, 0, lane),
                       load_f16pairs(qcol, 0, lane), sacc);
    sacc = wmma_f16f32(load_f16pairs(krow, 32, lane),
                       load_f16pairs(qcol, 32, lane), sacc);
    __syncthreads();                                   // protects bbuf/kbuf reuse
  }

  // store scores [blk][batch][h][i] with +bq and padding mask
  const int h = nl;
  if (h < NH_) {
    const float bqh = bq[(size_t)blk * NH_ + h];
#pragma unroll
    for (int r = 0; r < 8; ++r) {
      const int row = row0 + r + rbase;
      const int batch = row / I_;
      const int i = row - batch * I_;
      float s = sacc[r] + bqh;
      if (sga[row] == 0) s = MASKV_;
      scores[(((size_t)blk * B_ + batch) * NH_ + h) * I_ + i] = s;
    }
  }
}

// =====================================================================
// softmax over genes per head + sum over heads -> attn_wt (written to d_out)
//   grid (B, NB), block 256
// =====================================================================
__global__ void __launch_bounds__(256)
softmax_attn(const float* __restrict__ scores, float* __restrict__ attn_out) {
  const int b = blockIdx.x, blk = blockIdx.y, tid = threadIdx.x;
  const float* sc = scores + ((size_t)blk * B_ + b) * NH_ * I_;
  __shared__ float red[256];
  __shared__ float mxs[NH_], inv[NH_];

  for (int h = 0; h < NH_; ++h) {
    const float* s = sc + (size_t)h * I_;
    float m = -3.0e38f;
    for (int i = tid; i < I_; i += 256) m = fmaxf(m, s[i]);
    red[tid] = m; __syncthreads();
    for (int off = 128; off > 0; off >>= 1) {
      if (tid < off) red[tid] = fmaxf(red[tid], red[tid + off]);
      __syncthreads();
    }
    if (tid == 0) mxs[h] = red[0];
    __syncthreads();
    const float mh = mxs[h];
    float sum = 0.0f;
    for (int i = tid; i < I_; i += 256) sum += expf(s[i] - mh);
    red[tid] = sum; __syncthreads();
    for (int off = 128; off > 0; off >>= 1) {
      if (tid < off) red[tid] += red[tid + off];
      __syncthreads();
    }
    if (tid == 0) inv[h] = 1.0f / red[0];
    __syncthreads();
  }

  float* ao = attn_out + ((size_t)blk * B_ + b) * I_;
  for (int i = tid; i < I_; i += 256) {
    float w = 0.0f;
#pragma unroll
    for (int h = 0; h < NH_; ++h)
      w += expf(sc[(size_t)h * I_ + i] - mxs[h]) * inv[h];
    ao[i] = w;
  }
}

// =====================================================================
// weighted embedding sum: emb[blk][b][e] = sum_i attn[b][i] * table[g][e]
//   grid (B, NB), block 256 (2 fp32 cols per thread)
// =====================================================================
__global__ void __launch_bounds__(256)
weighted_emb(const float* __restrict__ emb_tables, const int* __restrict__ sga,
             const float* __restrict__ attn, float* __restrict__ emb_blk) {
  const int b = blockIdx.x, blk = blockIdx.y, tid = threadIdx.x;
  const float* tab = emb_tables + (size_t)blk * (V_ + 1) * E_;
  const float* w   = attn + ((size_t)blk * B_ + b) * I_;
  const int*   sg  = sga + (size_t)b * I_;
  const int e = tid * 2;
  float a0 = 0.0f, a1 = 0.0f;
  for (int i = 0; i < I_; ++i) {
    const int g = sg[i];
    const float wi = w[i];
    const float2 f = *reinterpret_cast<const float2*>(tab + (size_t)g * E_ + e);
    a0 += wi * f.x;
    a1 += wi * f.y;
  }
  float* o = emb_blk + ((size_t)blk * B_ + b) * E_ + e;
  o[0] = a0;
  o[1] = a1;
}

// =====================================================================
// block 0 combine: curr = relu(emb0 + can_emb[can]);  also hiddens[0]
// =====================================================================
__global__ void combine0(const float* __restrict__ emb0, const int* __restrict__ can,
                         const float* __restrict__ can_emb, float* __restrict__ curr,
                         float* __restrict__ hid_out) {
  const int idx = blockIdx.x * blockDim.x + threadIdx.x;   // B*E exact
  const int b = idx / E_, e = idx - b * E_;
  float v = emb0[idx] + can_emb[(size_t)can[b] * E_ + e];
  v = fmaxf(v, 0.0f);
  curr[idx] = v;
  hid_out[idx] = v;
}

// =====================================================================
// residual GEMM: out = act(curr_in @ Wh + emb_blk)   M=128,N=512,K=512
//   grid (2, 8), block 128; ACT 0=relu, 1=sigmoid
// =====================================================================
template <int ACT>
__global__ void __launch_bounds__(128)
res_gemm_kernel(const float* __restrict__ curr_in, const _Float16* __restrict__ WhTb,
                const float* __restrict__ embb, float* __restrict__ curr_out,
                float* __restrict__ hid_out) {
  const int lane = threadIdx.x & 31;
  const int wave = threadIdx.x >> 5;
  const int m0 = blockIdx.x * 64 + wave * 16;
  const int n0 = blockIdx.y * 64;
  const int nl = lane & 15;
  const float* rowbase = curr_in + (size_t)(m0 + nl) * E_;
  const _Float16* cb[4];
#pragma unroll
  for (int nf = 0; nf < 4; ++nf)
    cb[nf] = WhTb + (size_t)(n0 + nf * 16 + nl) * E_;

  v8f acc[4];
#pragma unroll
  for (int nf = 0; nf < 4; ++nf) { v8f z = {}; acc[nf] = z; }

  for (int ks = 0; ks < 16; ++ks) {
    const int k0 = ks * 32;
    const v16h a = load_a_f32(rowbase, k0, lane);
#pragma unroll
    for (int nf = 0; nf < 4; ++nf)
      acc[nf] = wmma_f16f32(a, load_f16pairs(cb[nf], k0, lane), acc[nf]);
  }

  const int rbase = (lane & 16) ? 8 : 0;
#pragma unroll
  for (int nf = 0; nf < 4; ++nf) {
    const int n = n0 + nf * 16 + nl;
#pragma unroll
    for (int r = 0; r < 8; ++r) {
      const int m = m0 + r + rbase;
      float v = acc[nf][r] + embb[(size_t)m * E_ + n];
      v = (ACT == 0) ? fmaxf(v, 0.0f) : (1.0f / (1.0f + expf(-v)));
      curr_out[(size_t)m * E_ + n] = v;
      hid_out[(size_t)m * E_ + n] = v;
    }
  }
}

// =====================================================================
// final GEMM: preds = curr @ Wfinal   M=128, N=5000 (padded 5056), K=512
//   grid (2, 79), block 128
// =====================================================================
__global__ void __launch_bounds__(128)
final_gemm_kernel(const float* __restrict__ curr_in, const _Float16* __restrict__ WfT,
                  float* __restrict__ preds) {
  const int lane = threadIdx.x & 31;
  const int wave = threadIdx.x >> 5;
  const int m0 = blockIdx.x * 64 + wave * 16;
  const int n0 = blockIdx.y * 64;
  const int nl = lane & 15;
  const float* rowbase = curr_in + (size_t)(m0 + nl) * E_;
  const _Float16* cb[4];
#pragma unroll
  for (int nf = 0; nf < 4; ++nf)
    cb[nf] = WfT + (size_t)(n0 + nf * 16 + nl) * E_;

  v8f acc[4];
#pragma unroll
  for (int nf = 0; nf < 4; ++nf) { v8f z = {}; acc[nf] = z; }

  for (int ks = 0; ks < 16; ++ks) {
    const int k0 = ks * 32;
    const v16h a = load_a_f32(rowbase, k0, lane);
#pragma unroll
    for (int nf = 0; nf < 4; ++nf)
      acc[nf] = wmma_f16f32(a, load_f16pairs(cb[nf], k0, lane), acc[nf]);
  }

  const int rbase = (lane & 16) ? 8 : 0;
#pragma unroll
  for (int nf = 0; nf < 4; ++nf) {
    const int n = n0 + nf * 16 + nl;
    if (n < OUT_) {
#pragma unroll
      for (int r = 0; r < 8; ++r) {
        const int m = m0 + r + rbase;
        preds[(size_t)m * OUT_ + n] = acc[nf][r];
      }
    }
  }
}

// =====================================================================
// launch
// =====================================================================
extern "C" void kernel_launch(void* const* d_in, const int* in_sizes, int n_in,
                              void* d_out, int out_size, void* d_ws, size_t ws_size,
                              hipStream_t stream) {
  (void)in_sizes; (void)n_in; (void)out_size; (void)ws_size;

  const int*   sga        = (const int*)d_in[0];
  const int*   can        = (const int*)d_in[1];
  const float* emb_tables = (const float*)d_in[2];
  const float* Wk         = (const float*)d_in[3];
  const float* bk         = (const float*)d_in[4];
  const float* Wq         = (const float*)d_in[5];
  const float* bq         = (const float*)d_in[6];
  const float* can_emb    = (const float*)d_in[7];
  const float* Wh         = (const float*)d_in[8];
  const float* Wfinal     = (const float*)d_in[9];

  float* out   = (float*)d_out;
  float* preds = out;                                   // [B, OUT]
  float* attns = out + (size_t)B_ * OUT_;               // [NB, B, I]
  float* hidds = attns + (size_t)NB_ * B_ * I_;         // [NB, B, E]

  char* ws = (char*)d_ws;
  size_t o = 0;
  _Float16* WkT = (_Float16*)(ws + o); o += (size_t)NB_ * E_ * E_ * 2;
  _Float16* WhT = (_Float16*)(ws + o); o += (size_t)(NB_ - 1) * E_ * E_ * 2;
  _Float16* WfT = (_Float16*)(ws + o); o += (size_t)NPAD_ * E_ * 2;
  _Float16* WqT = (_Float16*)(ws + o); o += (size_t)NB_ * 16 * E_ * 2;
  float* scores  = (float*)(ws + o);   o += (size_t)NB_ * B_ * NH_ * I_ * 4;
  float* emb_blk = (float*)(ws + o);   o += (size_t)NB_ * B_ * E_ * 4;
  float* currA   = (float*)(ws + o);   o += (size_t)B_ * E_ * 4;
  float* currB   = (float*)(ws + o);   o += (size_t)B_ * E_ * 4;

  // 1) weight prep (transpose + f16 convert)
  prep_wkT<<<(NB_ * E_ * E_ + 255) / 256, 256, 0, stream>>>(Wk, WkT);
  prep_whT<<<((NB_ - 1) * E_ * E_ + 255) / 256, 256, 0, stream>>>(Wh, WhT);
  prep_wfT<<<(NPAD_ * E_ + 255) / 256, 256, 0, stream>>>(Wfinal, WfT);
  prep_wqT<<<(NB_ * 16 * E_ + 255) / 256, 256, 0, stream>>>(Wq, WqT);

  // 2) fused key/score GEMM for all 4 blocks (independent)
  scores_kernel<<<dim3((B_ * I_) / 64, 1, NB_), 128, 0, stream>>>(
      sga, emb_tables, WkT, bk, WqT, bq, scores);

  // 3) softmax + head-sum -> attn weights (directly to d_out attns)
  softmax_attn<<<dim3(B_, NB_), 256, 0, stream>>>(scores, attns);

  // 4) attention-weighted embedding sums
  weighted_emb<<<dim3(B_, NB_), 256, 0, stream>>>(emb_tables, sga, attns, emb_blk);

  // 5) residual chain
  combine0<<<(B_ * E_) / 256, 256, 0, stream>>>(emb_blk, can, can_emb, currA, hidds);
  res_gemm_kernel<0><<<dim3(2, 8), 128, 0, stream>>>(
      currA, WhT, emb_blk + (size_t)1 * B_ * E_, currB, hidds + (size_t)1 * B_ * E_);
  res_gemm_kernel<0><<<dim3(2, 8), 128, 0, stream>>>(
      currB, WhT + (size_t)1 * E_ * E_, emb_blk + (size_t)2 * B_ * E_, currA,
      hidds + (size_t)2 * B_ * E_);
  res_gemm_kernel<1><<<dim3(2, 8), 128, 0, stream>>>(
      currA, WhT + (size_t)2 * E_ * E_, emb_blk + (size_t)3 * B_ * E_, currB,
      hidds + (size_t)3 * B_ * E_);

  // 6) final projection
  final_gemm_kernel<<<dim3(2, 79), 128, 0, stream>>>(currB, WfT, preds);
}
